// SwinTransformerBlock_41188736368745
// MI455X (gfx1250) — matrix operand order repaired and checked by reading, not compiled, not run on hardware
//
#include <hip/hip_runtime.h>
#include <hip/hip_bf16.h>
#include <stdint.h>

typedef __attribute__((ext_vector_type(16))) _Float16 v16h;
typedef __attribute__((ext_vector_type(8)))  float    v8f;

// ---- problem constants ----
constexpr int Cc     = 384;
constexpr int NHn    = 12;
constexpr int HEADd  = 32;
constexpr int SSs    = 4;
constexpr int Nn     = 64;            // tokens per window
constexpr int Bb     = 32;            // batch
constexpr int TOK    = Bb * 32 * 32;  // 32768 tokens
constexpr int WTt    = Bb * 16;       // 512 windows total
constexpr int HID    = 4 * Cc;        // 1536
constexpr float SCALEf = 0.17677669529663687f; // 32^-0.5

__device__ __forceinline__ v8f zero8() {
  v8f z;
  #pragma unroll
  for (int i = 0; i < 8; ++i) z[i] = 0.0f;
  return z;
}

// async global->LDS 16B copy (CDNA5 TDM-adjacent path, ASYNCcnt-tracked)
__device__ __forceinline__ void async_b128(uint32_t lds_off, uint64_t gaddr) {
  asm volatile("global_load_async_to_lds_b128 %0, %1, off"
               :: "v"(lds_off), "v"(gaddr) : "memory");
}
__device__ __forceinline__ void wait_async0() {
  asm volatile("s_wait_asynccnt 0x0" ::: "memory");
}
__device__ __forceinline__ void wait_async3() {
  asm volatile("s_wait_asynccnt 0x3" ::: "memory");
}

// ---------------- f32 -> f16 weight conversion ----------------
__global__ void cvt_f16(const float* __restrict__ s, _Float16* __restrict__ d, int n) {
  int i = blockIdx.x * 256 + threadIdx.x;
  if (i < n) d[i] = (_Float16)s[i];
}

// ---------------- LayerNorm (optionally fused shift+window-partition) ---------
__global__ __launch_bounds__(128) void ln_kernel(const float* __restrict__ x,
                                                 const float* __restrict__ g,
                                                 const float* __restrict__ bta,
                                                 _Float16* __restrict__ out,
                                                 int shifted) {
  int m = blockIdx.x;
  int tid = threadIdx.x;
  size_t src;
  if (shifted) {
    int b = m >> 10, r = m & 1023;
    int wi = r >> 6, t = r & 63;
    int wh = wi >> 2, ww = wi & 3;
    int ii = t >> 3, jj = t & 7;
    int sh = (wh * 8 + ii + SSs) & 31;   // roll(-SS): y[h] = x[(h+SS)%32]
    int sw = (ww * 8 + jj + SSs) & 31;
    src = ((size_t)(b << 10) + (sh << 5) + sw) * Cc;
  } else {
    src = (size_t)m * Cc;
  }
  float v0 = x[src + tid], v1 = x[src + tid + 128], v2 = x[src + tid + 256];
  __shared__ float red[128];
  red[tid] = v0 + v1 + v2;
  __syncthreads();
  #pragma unroll
  for (int off = 64; off > 0; off >>= 1) {
    if (tid < off) red[tid] += red[tid + off];
    __syncthreads();
  }
  float mu = red[0] * (1.0f / Cc);
  __syncthreads();
  float d0 = v0 - mu, d1 = v1 - mu, d2 = v2 - mu;
  red[tid] = d0 * d0 + d1 * d1 + d2 * d2;
  __syncthreads();
  #pragma unroll
  for (int off = 64; off > 0; off >>= 1) {
    if (tid < off) red[tid] += red[tid + off];
    __syncthreads();
  }
  float inv = rsqrtf(red[0] * (1.0f / Cc) + 1e-5f);
  size_t dst = (size_t)m * Cc;
  out[dst + tid]       = (_Float16)(d0 * inv * g[tid]       + bta[tid]);
  out[dst + tid + 128] = (_Float16)(d1 * inv * g[tid + 128] + bta[tid + 128]);
  out[dst + tid + 256] = (_Float16)(d2 * inv * g[tid + 256] + bta[tid + 256]);
}

// ---------------- generic f16 WMMA GEMM, async-LDS double-buffered ----------
struct GemmP {
  const _Float16* A;    // [M,K] row-major f16
  const _Float16* W;    // [K,N] row-major f16
  int M, K, N;
  const float* bias;    // [N]
  const float* addsrc;  // residual source (f32) for proj / fc2
  float* out32;         // f32 output for proj / fc2
  _Float16* o16a;       // q / fc1-hidden
  _Float16* o16b;       // k
  _Float16* o16c;       // v
  float scale;          // q scale
};

template <int EPI>
__global__ __launch_bounds__(256) void gemm_f16(GemmP p) {
  const int tid  = threadIdx.x;
  const int lane = tid & 31;
  const int wv   = tid >> 5;       // 8 waves
  const int wm   = wv & 1;         // 2 waves along M
  const int wn   = wv >> 1;        // 4 waves along N
  const int bm   = blockIdx.y * 64;
  const int bn   = blockIdx.x * 128;

  __shared__ alignas(16) _Float16 As[2][64][32];    // 2 x 4KB
  __shared__ alignas(16) _Float16 Bs[2][32][128];   // 2 x 8KB

  const uint64_t Ag = (uint64_t)(uintptr_t)p.A;
  const uint64_t Bg = (uint64_t)(uintptr_t)p.W;
  const uint32_t AsOff = (uint32_t)(uintptr_t)&As[0][0][0];
  const uint32_t BsOff = (uint32_t)(uintptr_t)&Bs[0][0][0];

  // issue one K-tile (A 64x32, B 32x128) into LDS buffer `buf`: 3 async b128/thread
  auto issue_tiles = [&](int k0, int buf) {
    {
      int r = tid >> 2, cq = tid & 3;                       // A: 64 rows x 4 chunks
      uint64_t ga = Ag + ((size_t)(bm + r) * p.K + k0) * 2 + cq * 16;
      async_b128(AsOff + buf * 4096 + r * 64 + cq * 16, ga);
    }
    #pragma unroll
    for (int j = 0; j < 2; ++j) {                           // B: 32 rows x 16 chunks
      int c = tid + j * 256;
      int r = c >> 4, cq = c & 15;
      uint64_t ga = Bg + ((size_t)(k0 + r) * p.N + bn) * 2 + cq * 16;
      async_b128(BsOff + buf * 8192 + r * 256 + cq * 16, ga);
    }
  };

  v8f acc[2][2];
  #pragma unroll
  for (int i = 0; i < 2; ++i)
    #pragma unroll
    for (int j = 0; j < 2; ++j) acc[i][j] = zero8();

  const int mrow  = lane & 15;
  const int koffA = (lane >> 4) * 8;   // A: lanes 0-15 -> K 0-7/16-23, 16-31 -> K 8-15/24-31
  const int koffB = (lane >> 4) * 16;  // B: same K order as A (lanes 0-15 -> K 0-7/16-23, 16-31 -> K 8-15/24-31)

  issue_tiles(0, 0);
  const int KT = p.K / 32;
  for (int kt = 0; kt < KT; ++kt) {
    const int buf = kt & 1;
    if (kt + 1 < KT) {
      issue_tiles((kt + 1) * 32, buf ^ 1);
      wait_async3();        // 3 next-tile loads may stay in flight; current tile landed
    } else {
      wait_async0();
    }
    __syncthreads();

    v16h afr[2], bfr[2];
    #pragma unroll
    for (int sm = 0; sm < 2; ++sm) {
      int m = wm * 32 + sm * 16 + mrow;
      #pragma unroll
      for (int t = 0; t < 8; ++t) {
        afr[sm][t]     = As[buf][m][koffA + t];
        afr[sm][8 + t] = As[buf][m][16 + koffA + t];
      }
    }
    #pragma unroll
    for (int sn = 0; sn < 2; ++sn) {
      int n = wn * 32 + sn * 16 + mrow;
      #pragma unroll
      for (int t = 0; t < 16; ++t) bfr[sn][t] = Bs[buf][koffA + t + (t >= 8 ? 8 : 0)][n];
    }
    #pragma unroll
    for (int sm = 0; sm < 2; ++sm)
      #pragma unroll
      for (int sn = 0; sn < 2; ++sn)
        acc[sm][sn] = __builtin_amdgcn_wmma_f32_16x16x32_f16(
            false, afr[sm], false, bfr[sn], (short)0, acc[sm][sn], false, false);
    __syncthreads();
  }

  // ---- epilogues ----
  #pragma unroll
  for (int sm = 0; sm < 2; ++sm) {
    #pragma unroll
    for (int sn = 0; sn < 2; ++sn) {
      int m0 = bm + wm * 32 + sm * 16;
      int n0 = bn + wn * 32 + sn * 16;
      #pragma unroll
      for (int r = 0; r < 8; ++r) {
        int row = m0 + r + ((lane >> 4) << 3);
        int col = n0 + (lane & 15);
        float val = acc[sm][sn][r] + p.bias[col];
        if constexpr (EPI == 0) {            // QKV -> scatter to [w,h,tok,d], scale q
          int s  = col / Cc;
          int cc = col % Cc;
          int hh = cc >> 5, dd = cc & 31;
          if (s == 0) val *= p.scale;
          size_t dst = ((size_t)((row >> 6) * NHn + hh) * Nn + (row & 63)) * HEADd + dd;
          _Float16* o = (s == 0) ? p.o16a : (s == 1) ? p.o16b : p.o16c;
          o[dst] = (_Float16)val;
        } else if constexpr (EPI == 1) {     // proj -> window-reverse + unshift + residual
          int b = row >> 10, rr = row & 1023;
          int wi = rr >> 6, t = rr & 63;
          int wh = wi >> 2, ww = wi & 3;
          int ii = t >> 3, jj = t & 7;
          int hh = (wh * 8 + ii + SSs) & 31;   // roll(+SS)
          int wc = (ww * 8 + jj + SSs) & 31;
          size_t ti = ((size_t)(b << 10) + (hh << 5) + wc) * Cc + col;
          p.out32[ti] = p.addsrc[ti] + val;
        } else if constexpr (EPI == 2) {     // fc1 -> tanh GELU -> f16
          float u = val;
          float gl = 0.5f * u * (1.0f + tanhf(0.79788456080286535588f *
                                              (u + 0.044715f * u * u * u)));
          p.o16a[(size_t)row * p.N + col] = (_Float16)gl;
        } else {                             // fc2 -> residual f32
          size_t ti = (size_t)row * p.N + col;
          p.out32[ti] = p.addsrc[ti] + val;
        }
      }
    }
  }
}

// ---------------- windowed attention: one block per (window, head) ----------
__device__ __forceinline__ int reg3(int x) { return x < 24 ? 0 : (x < 28 ? 1 : 2); }

__global__ __launch_bounds__(256) void attn_kernel(const _Float16* __restrict__ Q,
                                                   const _Float16* __restrict__ K,
                                                   const _Float16* __restrict__ V,
                                                   const float* __restrict__ bt,
                                                   _Float16* __restrict__ O) {
  const int tid = threadIdx.x, lane = tid & 31, wv = tid >> 5;
  const int bi = blockIdx.x;
  const int w = bi / NHn, h = bi % NHn;
  const int widx = w & 15, wh = widx >> 2, wwn = widx & 3;

  __shared__ alignas(16) _Float16 Qs[64][32], Ks[64][32], Vs[64][32];
  __shared__ float Ss[64][64];
  __shared__ _Float16 Ps[64][64];

  // stage q/k/v tiles via async global->LDS (3 x b128 per thread)
  {
    size_t base = (((size_t)w * NHn + h) * Nn * HEADd) * 2;   // bytes
    int r = tid >> 2, cq = tid & 3;                           // 64 rows x 4 chunks
    uint32_t loff = (uint32_t)(r * 64 + cq * 16);
    uint64_t goff = base + r * 64 + cq * 16;
    async_b128((uint32_t)(uintptr_t)&Qs[0][0] + loff, (uint64_t)(uintptr_t)Q + goff);
    async_b128((uint32_t)(uintptr_t)&Ks[0][0] + loff, (uint64_t)(uintptr_t)K + goff);
    async_b128((uint32_t)(uintptr_t)&Vs[0][0] + loff, (uint64_t)(uintptr_t)V + goff);
    wait_async0();
  }
  __syncthreads();

  const int mrow  = lane & 15;
  const int koffA = (lane >> 4) * 8;
  const int koffB = (lane >> 4) * 16;

  // S = q * k^T  (16 tiles of 16x16, K=32 = single WMMA each; 2 per wave)
  #pragma unroll
  for (int it = 0; it < 2; ++it) {
    int tile = wv * 2 + it;
    int m0 = (tile >> 2) * 16, n0 = (tile & 3) * 16;
    v16h a, bfr;
    int m = m0 + mrow, n = n0 + mrow;
    #pragma unroll
    for (int t = 0; t < 8; ++t) { a[t] = Qs[m][koffA + t]; a[8 + t] = Qs[m][16 + koffA + t]; }
    #pragma unroll
    for (int t = 0; t < 16; ++t) bfr[t] = Ks[n][koffA + t + (t >= 8 ? 8 : 0)];   // (k^T)[kd][n] = Ks[n][kd]
    v8f c = zero8();
    c = __builtin_amdgcn_wmma_f32_16x16x32_f16(false, a, false, bfr, (short)0, c, false, false);
    #pragma unroll
    for (int r = 0; r < 8; ++r) {
      int row = m0 + r + ((lane >> 4) << 3);
      int col = n0 + (lane & 15);
      int i1 = row >> 3, j1 = row & 7, i2 = col >> 3, j2 = col & 7;
      float bias = bt[((i1 - i2 + 7) * 15 + (j1 - j2 + 7)) * NHn + h];
      int la = reg3(wh * 8 + i1) * 3 + reg3(wwn * 8 + j1);
      int lb = reg3(wh * 8 + i2) * 3 + reg3(wwn * 8 + j2);
      float msk = (la != lb) ? -100.0f : 0.0f;
      Ss[row][col] = c[r] + bias + msk;
    }
  }
  __syncthreads();

  // row softmax (f32)
  if (tid < 64) {
    float mx = -1e30f;
    #pragma unroll 4
    for (int c2 = 0; c2 < 64; ++c2) mx = fmaxf(mx, Ss[tid][c2]);
    float sum = 0.0f;
    #pragma unroll 4
    for (int c2 = 0; c2 < 64; ++c2) { float e = __expf(Ss[tid][c2] - mx); Ss[tid][c2] = e; sum += e; }
    float inv = 1.0f / sum;
    #pragma unroll 4
    for (int c2 = 0; c2 < 64; ++c2) Ps[tid][c2] = (_Float16)(Ss[tid][c2] * inv);
  }
  __syncthreads();

  // O = P * V  (8 tiles, K=64 -> 2 chained WMMAs; 1 tile per wave)
  {
    int m0 = (wv >> 1) * 16, n0 = (wv & 1) * 16;
    v8f acc = zero8();
    #pragma unroll
    for (int kk = 0; kk < 64; kk += 32) {
      v16h a, bfr;
      int m = m0 + mrow, n = n0 + mrow;
      #pragma unroll
      for (int t = 0; t < 8; ++t) { a[t] = Ps[m][kk + koffA + t]; a[8 + t] = Ps[m][kk + 16 + koffA + t]; }
      #pragma unroll
      for (int t = 0; t < 16; ++t) bfr[t] = Vs[kk + koffA + t + (t >= 8 ? 8 : 0)][n];
      acc = __builtin_amdgcn_wmma_f32_16x16x32_f16(false, a, false, bfr, (short)0, acc, false, false);
    }
    #pragma unroll
    for (int r = 0; r < 8; ++r) {
      int row = m0 + r + ((lane >> 4) << 3);
      int col = n0 + (lane & 15);
      O[((size_t)w * Nn + row) * Cc + h * HEADd + col] = (_Float16)acc[r];
    }
  }
}

// ---------------- host side ----------------
extern "C" void kernel_launch(void* const* d_in, const int* in_sizes, int n_in,
                              void* d_out, int out_size, void* d_ws, size_t ws_size,
                              hipStream_t stream) {
  const float* x        = (const float*)d_in[0];
  const float* n1g      = (const float*)d_in[1];
  const float* n1b      = (const float*)d_in[2];
  const float* qkv_w    = (const float*)d_in[3];
  const float* qkv_b    = (const float*)d_in[4];
  const float* rel_bt   = (const float*)d_in[5];
  const float* proj_w   = (const float*)d_in[6];
  const float* proj_b   = (const float*)d_in[7];
  const float* n2g      = (const float*)d_in[8];
  const float* n2b      = (const float*)d_in[9];
  const float* fc1_w    = (const float*)d_in[10];
  const float* fc1_b    = (const float*)d_in[11];
  const float* fc2_w    = (const float*)d_in[12];
  const float* fc2_b    = (const float*)d_in[13];
  float* out = (float*)d_out;

  uint8_t* wsp = (uint8_t*)d_ws;
  auto alloc = [&](size_t bytes) -> void* {
    void* p = (void*)wsp;
    wsp += (bytes + 255) & ~(size_t)255;
    return p;
  };
  _Float16* wq16  = (_Float16*)alloc((size_t)Cc * 3 * Cc * 2);
  _Float16* wp16  = (_Float16*)alloc((size_t)Cc * Cc * 2);
  _Float16* w116  = (_Float16*)alloc((size_t)Cc * HID * 2);
  _Float16* w216  = (_Float16*)alloc((size_t)HID * Cc * 2);
  _Float16* winA  = (_Float16*)alloc((size_t)TOK * Cc * 2);   // reused as LN2 output
  _Float16* q16   = (_Float16*)alloc((size_t)TOK * Cc * 2);
  _Float16* k16   = (_Float16*)alloc((size_t)TOK * Cc * 2);
  _Float16* v16   = (_Float16*)alloc((size_t)TOK * Cc * 2);
  _Float16* ao16  = (_Float16*)alloc((size_t)TOK * Cc * 2);
  float*    x2    = (float*)   alloc((size_t)TOK * Cc * 4);
  _Float16* h1    = (_Float16*)alloc((size_t)TOK * HID * 2);

  auto cvt = [&](const float* s, _Float16* d, int n) {
    cvt_f16<<<(n + 255) / 256, 256, 0, stream>>>(s, d, n);
  };
  cvt(qkv_w, wq16, Cc * 3 * Cc);
  cvt(proj_w, wp16, Cc * Cc);
  cvt(fc1_w, w116, Cc * HID);
  cvt(fc2_w, w216, HID * Cc);

  // LN1 + shift + window partition
  ln_kernel<<<TOK, 128, 0, stream>>>(x, n1g, n1b, winA, 1);

  // QKV GEMM
  {
    GemmP p{};
    p.A = winA; p.W = wq16; p.M = TOK; p.K = Cc; p.N = 3 * Cc;
    p.bias = qkv_b; p.o16a = q16; p.o16b = k16; p.o16c = v16; p.scale = SCALEf;
    gemm_f16<0><<<dim3((3 * Cc) / 128, TOK / 64), 256, 0, stream>>>(p);
  }

  // attention per (window, head)
  attn_kernel<<<WTt * NHn, 256, 0, stream>>>(q16, k16, v16, rel_bt, ao16);

  // proj GEMM + window reverse + unshift + residual
  {
    GemmP p{};
    p.A = ao16; p.W = wp16; p.M = TOK; p.K = Cc; p.N = Cc;
    p.bias = proj_b; p.addsrc = x; p.out32 = x2;
    gemm_f16<1><<<dim3(Cc / 128, TOK / 64), 256, 0, stream>>>(p);
  }

  // LN2 (plain token order) -> reuse winA buffer
  ln_kernel<<<TOK, 128, 0, stream>>>(x2, n2g, n2b, winA, 0);

  // FC1 + GELU
  {
    GemmP p{};
    p.A = winA; p.W = w116; p.M = TOK; p.K = Cc; p.N = HID;
    p.bias = fc1_b; p.o16a = h1;
    gemm_f16<2><<<dim3(HID / 128, TOK / 64), 256, 0, stream>>>(p);
  }

  // FC2 + residual -> output
  {
    GemmP p{};
    p.A = h1; p.W = w216; p.M = TOK; p.K = HID; p.N = Cc;
    p.bias = fc2_b; p.addsrc = x2; p.out32 = out;
    gemm_f16<3><<<dim3(Cc / 128, TOK / 64), 256, 0, stream>>>(p);
  }

  (void)in_sizes; (void)n_in; (void)out_size; (void)ws_size;
}
